// MultiHeadAttention_25967372272324
// MI455X (gfx1250) — matrix operand order repaired and checked by reading, not compiled, not run on hardware
//
#include <hip/hip_runtime.h>

// ---------------------------------------------------------------------------
// MHA forward for MI455X (gfx1250, wave32, WMMA).
// B=2, S=4096, D=512, H=8, depth=64. bf16 WMMA w/ fp32 accumulate.
// Flash attention with double-buffered TDM (tensor_load_to_lds / TENSORcnt)
// K/V staging in LDS; async-load fallback if the TDM builtin is absent.
// ---------------------------------------------------------------------------

typedef __attribute__((ext_vector_type(16))) __bf16 v16bf;
typedef __attribute__((ext_vector_type(8)))  float  v8f;
typedef __attribute__((ext_vector_type(4)))  unsigned tdm_g0;
typedef __attribute__((ext_vector_type(8)))  int      tdm_g1;
typedef __attribute__((ext_vector_type(4)))  int      tdm_g2;

#define SEQ   4096
#define DM    512
#define HEADS 8
#define DH    64

#if defined(__has_builtin)
#if __has_builtin(__builtin_amdgcn_tensor_load_to_lds)
#define HAVE_TDM 1
#endif
#endif
#ifndef HAVE_TDM
#define HAVE_TDM 0
#endif

static __device__ __forceinline__ v8f wmma32(v16bf a, v16bf b, v8f c) {
  // D = A(16x32 bf16) x B(32x16 bf16) + C(16x16 f32)
  return __builtin_amdgcn_wmma_f32_16x16x32_bf16(
      /*neg_a=*/false, a, /*neg_b=*/false, b,
      /*c_mod=*/(short)0, c, /*reuse_a=*/false, /*reuse_b=*/false);
}

// A-fragment (16-bit, 16x32): lane l (lo=l&15, hh=l>>4) holds row lo,
// K = {hh*8+0..7, 16+hh*8+0..7}.  Source is row-major [16, ld].
template <typename T>
static __device__ __forceinline__ v16bf load_a_frag(const T* base, int ld) {
  const int l  = threadIdx.x & 31;
  const int lo = l & 15;
  const int hh = l >> 4;
  const T* p = base + (size_t)lo * ld + hh * 8;
  v16bf r;
#pragma unroll
  for (int i = 0; i < 8; ++i) {
    r[i]     = (__bf16)(float)p[i];
    r[i + 8] = (__bf16)(float)p[i + 16];
  }
  return r;
}

// B-fragment (16-bit, 32x16): lane l holds column lo, K = {hh*8+0..7, 16+hh*8+0..7}.
// Source stored K-contiguous per column: row-major [16 cols, ld].
static __device__ __forceinline__ v16bf load_b_frag(const __bf16* base, int ld) {
  const int l  = threadIdx.x & 31;
  const int lo = l & 15;
  const int hh = l >> 4;
  const __bf16* p = base + (size_t)lo * ld + hh * 8;
  v16bf r;
#pragma unroll
  for (int i = 0; i < 8; ++i) {
    r[i]     = p[i];
    r[i + 8] = p[i + 16];
  }
  return r;
}

// ---- lane xor-exchange via ds_swizzle (group-of-32: xor in offset[14:10]) --
template <int IMM>
static __device__ __forceinline__ float swz(float v) {
  return __builtin_bit_cast(
      float, __builtin_amdgcn_ds_swizzle(__builtin_bit_cast(int, v), IMM));
}
static __device__ __forceinline__ float red_max16(float t) {
  t = fmaxf(t, swz<0x041F>(t));   // SWAPX1
  t = fmaxf(t, swz<0x081F>(t));   // SWAPX2
  t = fmaxf(t, swz<0x101F>(t));   // SWAPX4
  t = fmaxf(t, swz<0x201F>(t));   // SWAPX8
  return t;
}
static __device__ __forceinline__ float red_sum16(float t) {
  t += swz<0x041F>(t);
  t += swz<0x081F>(t);
  t += swz<0x101F>(t);
  t += swz<0x201F>(t);
  return t;
}

// ---- CDNA5 LDS staging paths ----------------------------------------------
static __device__ __forceinline__ unsigned lds_off(const void* p) {
  // generic pointer to LDS: low 32 bits are the wave-relative LDS byte addr
  return (unsigned)(uintptr_t)p;
}

#if HAVE_TDM
// One 2D TDM descriptor: tile_h rows x tile_w bf16 elements, global row
// stride `stride` elements, packed contiguously into LDS at lds_addr.
static __device__ __forceinline__ void tdm_load_2d(unsigned lds_addr,
                                                   const void* gaddr,
                                                   unsigned tile_w,
                                                   unsigned tile_h,
                                                   unsigned stride) {
  const unsigned long long ga = (unsigned long long)(uintptr_t)gaddr;
  tdm_g0 g0;
  g0[0] = 1u;                                   // count=1, user descriptor
  g0[1] = lds_addr;                             // lds_addr [63:32]
  g0[2] = (unsigned)ga;                         // global_addr [95:64]
  g0[3] = (unsigned)((ga >> 32) & 0x01FFFFFFu)  // global_addr [120:96]
          | 0x80000000u;                        // type=2 ("image") [127:126]
  tdm_g1 g1;
  g1[0] = (int)(1u << 16);                      // data_size=1 -> 2 bytes
  g1[1] = (int)(tile_w << 16);                  // tensor_dim0[15:0] @ [63:48]
  g1[2] = (int)(tile_h << 16);                  // tensor_dim1[15:0] @ [95:80]
  g1[3] = (int)(tile_w << 16);                  // tile_dim0 @ [127:112]
  g1[4] = (int)(tile_h & 0xFFFFu);              // tile_dim1 @ [143:128]
  g1[5] = (int)stride;                          // tensor_dim0_stride[31:0]
  g1[6] = 0;
  g1[7] = 0;
  tdm_g2 z4 = {0, 0, 0, 0};                     // 2D: groups 2/3 unused
  tdm_g1 z8 = {0, 0, 0, 0, 0, 0, 0, 0};
  // 6-arg toolchain form: (g0 u32x4, g1 i32x8, g2 i32x4, g3 i32x4, i32x8, cpol)
  __builtin_amdgcn_tensor_load_to_lds(g0, g1, z4, z4, z8, 0);
}
#else
static __device__ __forceinline__ void async_cp16(unsigned loff, const void* g) {
  asm volatile("global_load_async_to_lds_b128 %0, %1, off"
               :: "v"(loff), "v"((unsigned long long)(uintptr_t)g)
               : "memory");
}
static __device__ __forceinline__ void wait_async0() {
  asm volatile("s_wait_asynccnt 0" ::: "memory");
}
#endif

// ---------------------------------------------------------------------------
// Kernel 0: W [512,512] fp32 row-major (k,n) -> Wt [512,512] bf16 (n,k)
// ---------------------------------------------------------------------------
__global__ void wt_kernel(const float* __restrict__ W, __bf16* __restrict__ Wt) {
  const int e = blockIdx.x * blockDim.x + threadIdx.x;  // 0..262143
  const int k = e >> 9;
  const int n = e & 511;
  Wt[(size_t)n * DM + k] = (__bf16)W[e];  // coalesced read
}

// ---------------------------------------------------------------------------
// Kernel 1: GEMM  OUT[8192,512] = X[8192,512] @ W[512,512] + bias
// MODE 0: out bf16, split-heads [B*H, S, 64]          (Q, K)
// MODE 1: out bf16, split-heads transposed [B*H,64,S] (V)
// MODE 2: out fp32, [8192, 512]                       (final projection)
// ---------------------------------------------------------------------------
template <int MODE, typename TIn>
__global__ void gemm512_kernel(const TIn* __restrict__ X,
                               const __bf16* __restrict__ Wt,
                               const float* __restrict__ bias,
                               void* __restrict__ outp) {
  const int wave = threadIdx.x >> 5;
  const int l    = threadIdx.x & 31;
  const int lo   = l & 15;
  const int hh   = l >> 4;
  const int m0   = blockIdx.x * 16;
  const int n0   = (blockIdx.y * 8 + wave) * 16;

  const TIn*    xrow = X  + (size_t)m0 * DM;
  const __bf16* wrow = Wt + (size_t)n0 * DM;

  v8f c = {};
#pragma unroll 4
  for (int k0 = 0; k0 < DM; k0 += 32) {
    v16bf a = load_a_frag<TIn>(xrow + k0, DM);
    v16bf b = load_b_frag(wrow + k0, DM);
    c = wmma32(a, b, c);
  }

  const int   n  = n0 + lo;
  const float bv = bias[n];
#pragma unroll
  for (int r = 0; r < 8; ++r) c[r] += bv;

  if (MODE == 2) {
    float* out = (float*)outp;
#pragma unroll
    for (int r = 0; r < 8; ++r) {
      const int m = m0 + r + 8 * hh;
      out[(size_t)m * DM + n] = c[r];
    }
  } else {
    __bf16* out  = (__bf16*)outp;
    const int b_    = m0 >> 12;          // batch (tile never crosses batch)
    const int sbase = m0 & (SEQ - 1);
    const int head  = n >> 6;
    const int d     = n & 63;
#pragma unroll
    for (int r = 0; r < 8; ++r) {
      const int s = sbase + r + 8 * hh;
      if (MODE == 0)
        out[(((size_t)(b_ * HEADS + head) * SEQ) + s) * DH + d] = (__bf16)c[r];
      else
        out[(((size_t)(b_ * HEADS + head) * DH) + d) * SEQ + s] = (__bf16)c[r];
    }
  }
}

// ---------------------------------------------------------------------------
// Kernel 2: flash attention.  4 waves/block share one bh; each wave owns a
// 16-query tile.  32-key blocks of K [32,64] and Vt [64,32] are staged into
// LDS by the Tensor Data Mover (double buffered, TENSORcnt).
// Q,K: [B*H,S,64] bf16; Vt: [B*H,64,S] bf16.  Writes Oc [B,S,512] bf16.
// Grid (64, 16), block 128 (4 waves).
// ---------------------------------------------------------------------------
__global__ void flash_attn_kernel(const __bf16* __restrict__ Q,
                                  const __bf16* __restrict__ K,
                                  const __bf16* __restrict__ Vt,
                                  const float* __restrict__ mask,
                                  __bf16* __restrict__ Oc) {
  const int tid  = threadIdx.x;
  const int wave = tid >> 5;
  const int l    = tid & 31;
  const int lo   = l & 15;
  const int hh   = l >> 4;
  const int bh   = blockIdx.y;
  const int b    = bh >> 3;
  const int head = bh & 7;
  const int q0   = (blockIdx.x * 4 + wave) * 16;

  const __bf16* Qb = Q  + ((size_t)bh * SEQ + q0) * DH;
  const __bf16* Kb = K  + (size_t)bh * SEQ * DH;
  const __bf16* Vb = Vt + (size_t)bh * DH * SEQ;
  const float*  Mb = mask + (size_t)b * SEQ * SEQ + (size_t)q0 * SEQ;

  __shared__ __bf16 kbuf[2][32 * DH];   // K block  [32 keys][64 d]
  __shared__ __bf16 vbuf[2][DH * 32];   // Vt block [64 d][32 keys]
  __shared__ __bf16 pbuf[4][16 * 32];   // per-wave P relayout tile (C -> A)
  __bf16* pb = pbuf[wave];

  // cooperative stage of one 32-key block (4 KB K + 4 KB Vt)
  auto stage_kv = [&](int buf, int kb) {
#if HAVE_TDM
    if (wave == 0) {  // TDM ignores EXEC; issue each descriptor once
      tdm_load_2d(lds_off(kbuf[buf]), Kb + (size_t)kb * DH, DH, 32, DH);
      tdm_load_2d(lds_off(vbuf[buf]), Vb + kb, 32, DH, SEQ);
    }
#else
    const char* gk = (const char*)(Kb + (size_t)kb * DH);  // 4 KB contiguous
    char*       lk = (char*)kbuf[buf];
    for (int c = tid; c < 256; c += 128)
      async_cp16(lds_off(lk + c * 16), gk + c * 16);
    char* lv = (char*)vbuf[buf];
    for (int c = tid; c < 256; c += 128) {
      const int d = c >> 2, part = c & 3;
      async_cp16(lds_off(lv + d * 64 + part * 16),
                 Vb + (size_t)d * SEQ + kb + part * 8);
    }
#endif
  };
  auto wait_stage = [&]() {
#if HAVE_TDM
    if (wave == 0) __builtin_amdgcn_s_wait_tensorcnt(0);
#else
    wait_async0();
#endif
    __syncthreads();  // issuing wave(s) waited -> LDS block fully populated
  };

  // Q tile 16x64 = two A-fragments (K=0..31, 32..63), kept resident
  const v16bf aq0 = load_a_frag<__bf16>(Qb, DH);
  const v16bf aq1 = load_a_frag<__bf16>(Qb + 32, DH);

  v8f o0 = {}, o1 = {}, o2 = {}, o3 = {};  // O accumulator 16x64 f32
  float mrow[8], lrow[8];
#pragma unroll
  for (int r = 0; r < 8; ++r) { mrow[r] = -3.0e38f; lrow[r] = 0.f; }

  const float inv_sqrt_d = 0.125f;  // 1/sqrt(64)

  stage_kv(0, 0);
  int p = 0;

  for (int kb = 0; kb < SEQ; kb += 32) {
    wait_stage();

    if (kb + 32 < SEQ) stage_kv(p ^ 1, kb + 32);  // overlap next stage

    const __bf16* kl = kbuf[p];
    const __bf16* vl = vbuf[p];

    // ---- batched mask loads (one clause, overlaps score WMMAs) -----------
    float msk0[8], msk1[8];
#pragma unroll
    for (int r = 0; r < 8; ++r) {
      const int qr = r + 8 * hh;
      msk0[r] = Mb[(size_t)qr * SEQ + kb + lo];
      msk1[r] = Mb[(size_t)qr * SEQ + kb + 16 + lo];
    }
    if (kb + 32 < SEQ)  // prefetch next mask columns (global_prefetch_b8)
      __builtin_prefetch(Mb + (size_t)(8 * hh) * SEQ + kb + 32 + lo, 0, 1);

    // ---- scores: S[16 q, 32 k] as two 16x16 WMMA accumulations -----------
    v8f s0 = {}, s1 = {};
    s0 = wmma32(aq0, load_b_frag(kl, DH), s0);                 // keys 0..15
    s0 = wmma32(aq1, load_b_frag(kl + 32, DH), s0);
    s1 = wmma32(aq0, load_b_frag(kl + 16 * DH, DH), s1);       // keys 16..31
    s1 = wmma32(aq1, load_b_frag(kl + 16 * DH + 32, DH), s1);

    // ---- scale + mask + online softmax (row stats across 16-lane group) --
    float p0[8], p1[8], scale[8];
#pragma unroll
    for (int r = 0; r < 8; ++r) {
      const float v0 = s0[r] * inv_sqrt_d + msk0[r] * 1e-9f;
      const float v1 = s1[r] * inv_sqrt_d + msk1[r] * 1e-9f;
      const float t    = red_max16(fmaxf(v0, v1));
      const float mnew = fmaxf(mrow[r], t);
      scale[r] = __expf(mrow[r] - mnew);
      mrow[r]  = mnew;
      p0[r] = __expf(v0 - mnew);
      p1[r] = __expf(v1 - mnew);
      lrow[r] = lrow[r] * scale[r] + red_sum16(p0[r] + p1[r]);
    }
#pragma unroll
    for (int r = 0; r < 8; ++r) {
      o0[r] *= scale[r]; o1[r] *= scale[r]; o2[r] *= scale[r]; o3[r] *= scale[r];
    }

    // ---- P: C-fragment layout -> row-major LDS -> A-fragment -------------
#pragma unroll
    for (int r = 0; r < 8; ++r) {
      const int qr = r + 8 * hh;
      pb[qr * 32 + lo]      = (__bf16)p0[r];
      pb[qr * 32 + 16 + lo] = (__bf16)p1[r];
    }
    asm volatile("s_wait_dscnt 0" ::: "memory");  // wave-local LDS RAW
    v16bf pa = load_a_frag<__bf16>(pb, 32);

    // ---- O += P(16x32) @ V(32x64): four d-tiles (B-frags from LDS) -------
    o0 = wmma32(pa, load_b_frag(vl + 0 * 32, 32), o0);
    o1 = wmma32(pa, load_b_frag(vl + 16 * 32, 32), o1);
    o2 = wmma32(pa, load_b_frag(vl + 32 * 32, 32), o2);
    o3 = wmma32(pa, load_b_frag(vl + 48 * 32, 32), o3);

    __syncthreads();      // all reads of buffer p done before it is restaged
    p ^= 1;
  }

  // ---- normalize by row sum, store concat [B,S,512] bf16 -----------------
#pragma unroll
  for (int r = 0; r < 8; ++r) {
    const float rl = 1.0f / lrow[r];
    const int   q  = q0 + r + 8 * hh;
    __bf16* orow = Oc + ((size_t)b * SEQ + q) * DM + head * DH;
    orow[lo]      = (__bf16)(o0[r] * rl);
    orow[16 + lo] = (__bf16)(o1[r] * rl);
    orow[32 + lo] = (__bf16)(o2[r] * rl);
    orow[48 + lo] = (__bf16)(o3[r] * rl);
  }
}

// ---------------------------------------------------------------------------
extern "C" void kernel_launch(void* const* d_in, const int* in_sizes, int n_in,
                              void* d_out, int out_size, void* d_ws, size_t ws_size,
                              hipStream_t stream) {
  (void)in_sizes; (void)n_in; (void)out_size; (void)ws_size;

  const float* query = (const float*)d_in[0];
  const float* key_  = (const float*)d_in[1];
  const float* value = (const float*)d_in[2];
  const float* mask  = (const float*)d_in[3];
  const float* Wq = (const float*)d_in[4];
  const float* bq = (const float*)d_in[5];
  const float* Wk = (const float*)d_in[6];
  const float* bk = (const float*)d_in[7];
  const float* Wv = (const float*)d_in[8];
  const float* bv = (const float*)d_in[9];
  const float* Wo = (const float*)d_in[10];
  const float* bo = (const float*)d_in[11];

  // workspace carve-up (bf16 everywhere internal)
  char* ws = (char*)d_ws;
  const size_t WSZ = (size_t)DM * DM * sizeof(__bf16);          // 512 KB
  const size_t TSZ = (size_t)16 * SEQ * DH * sizeof(__bf16);    // 8 MB
  size_t off = 0;
  __bf16* WqT = (__bf16*)(ws + off); off += WSZ;
  __bf16* WkT = (__bf16*)(ws + off); off += WSZ;
  __bf16* WvT = (__bf16*)(ws + off); off += WSZ;
  __bf16* WoT = (__bf16*)(ws + off); off += WSZ;
  __bf16* Qb  = (__bf16*)(ws + off); off += TSZ;
  __bf16* Kb  = (__bf16*)(ws + off); off += TSZ;
  __bf16* Vtb = (__bf16*)(ws + off); off += TSZ;
  __bf16* Ocb = (__bf16*)(ws + off); off += TSZ;

  // 0) weight transpose + bf16 convert
  wt_kernel<<<1024, 256, 0, stream>>>(Wq, WqT);
  wt_kernel<<<1024, 256, 0, stream>>>(Wk, WkT);
  wt_kernel<<<1024, 256, 0, stream>>>(Wv, WvT);
  wt_kernel<<<1024, 256, 0, stream>>>(Wo, WoT);

  // 1) Q/K/V projections (fused bias + split-heads + bf16 cast)
  dim3 gg(512, 4);
  gemm512_kernel<0, float><<<gg, 256, 0, stream>>>(query, WqT, bq, (void*)Qb);
  gemm512_kernel<0, float><<<gg, 256, 0, stream>>>(key_,  WkT, bk, (void*)Kb);
  gemm512_kernel<1, float><<<gg, 256, 0, stream>>>(value, WvT, bv, (void*)Vtb);

  // 2) flash attention -> concat
  flash_attn_kernel<<<dim3(64, 16), 128, 0, stream>>>(Qb, Kb, Vtb, mask, Ocb);

  // 3) output projection -> fp32 d_out
  gemm512_kernel<2, __bf16><<<gg, 256, 0, stream>>>(Ocb, WoT, bo, d_out);
}
